// InvariantPointAttention_83837761618294
// MI455X (gfx1250) — compile-verified, matrix operand order checked by reading, not run on hardware
//
#include <hip/hip_runtime.h>
#include <hip/hip_bf16.h>
#include <stdint.h>

typedef __attribute__((ext_vector_type(16))) _Float16 v16h;
typedef __attribute__((ext_vector_type(8)))  float    v8f;

#define B_  2
#define N_  2048
#define D_  256
#define H_  8
#define HD_ 32
#define OUTIN_ 456
#define KPAD_  480
#define BN_ (B_ * N_)

// ---------------------------------------------------------------------------
// WMMA helpers (CDNA5, wave32).  Layouts per ISA 7.12.2:
//   A (16x32 f16): lane L -> row M = L&15, K-halves: f[0..7] = K[ko..ko+7],
//                  f[8..15] = K[ko+16..ko+23] with ko = (L>>4)*8.
//   B (32x16 f16): mirrored, lane -> column N.
//   C/D (16x16 f32): vgpr r -> row r + 8*(L>>4), lane&15 -> column.
// ---------------------------------------------------------------------------
__device__ inline v8f wmma_f16(v16h a, v16h b, v8f c) {
  return __builtin_amdgcn_wmma_f32_16x16x32_f16(
      /*neg_a=*/false, a, /*neg_b=*/false, b,
      /*c_mod=*/(short)0, c, /*reuse_a=*/false, /*reuse_b=*/false);
}

__device__ inline v16h load_frag(const _Float16* __restrict__ base, int ldr) {
  int lane = threadIdx.x & 31;
  const _Float16* p = base + (long)(lane & 15) * ldr + ((lane >> 4) * 8);
  v16h f;
#pragma unroll
  for (int i = 0; i < 8; ++i) { f[i] = p[i]; f[i + 8] = p[16 + i]; }
  return f;
}

__device__ inline v16h load_frag_pred(const _Float16* __restrict__ base, int ldr, bool valid) {
  int lane = threadIdx.x & 31;
  v16h f;
  if (valid) {
    const _Float16* p = base + (long)(lane & 15) * ldr + ((lane >> 4) * 8);
#pragma unroll
    for (int i = 0; i < 8; ++i) { f[i] = p[i]; f[i + 8] = p[16 + i]; }
  } else {
#pragma unroll
    for (int i = 0; i < 16; ++i) f[i] = (_Float16)0.0f;
  }
  return f;
}

// ---------------------------------------------------------------------------
// Kernel 1: data prep (convert/transpose/pad/zero + head-weight softmax)
// ---------------------------------------------------------------------------
__global__ void ipa_prep(const float* __restrict__ x, const float* __restrict__ trans,
                         const float* __restrict__ Wq, const float* __restrict__ Wk,
                         const float* __restrict__ Wv, const float* __restrict__ Wo,
                         const float* __restrict__ hwin,
                         _Float16* __restrict__ x_h, _Float16* __restrict__ wt,
                         _Float16* __restrict__ wo_t, _Float16* __restrict__ tT,
                         _Float16* __restrict__ feat, float* __restrict__ hsoft) {
  long idx = (long)blockIdx.x * blockDim.x + threadIdx.x;
  switch (blockIdx.y) {
    case 0:  // x -> f16
      if (idx < (long)BN_ * D_) x_h[idx] = (_Float16)x[idx];
      break;
    case 1: case 2: case 3: {  // Wq/Wk/Wv transposed -> [n][k] f16
      const float* W = (blockIdx.y == 1) ? Wq : (blockIdx.y == 2) ? Wk : Wv;
      if (idx < (long)D_ * D_) {
        int n = (int)(idx >> 8), k = (int)(idx & 255);
        wt[(long)(blockIdx.y - 1) * D_ * D_ + idx] = (_Float16)W[(long)k * D_ + n];
      }
      break; }
    case 4:  // Wo transposed + K-padded -> [256][480] f16
      if (idx < (long)D_ * KPAD_) {
        int n = (int)(idx / KPAD_), k = (int)(idx % KPAD_);
        wo_t[idx] = (_Float16)((k < OUTIN_) ? Wo[(long)k * D_ + n] : 0.0f);
      }
      break;
    case 5:  // translations transposed -> [b][3][N] f16
      if (idx < (long)B_ * 3 * N_) {
        int b = (int)(idx / (3 * N_));
        int c = (int)((idx / N_) % 3);
        int j = (int)(idx % N_);
        tT[idx] = (_Float16)trans[((long)b * N_ + j) * 3 + c];
      }
      break;
    case 6:  // zero padded feature buffer
      if (idx < (long)BN_ * KPAD_) feat[idx] = (_Float16)0.0f;
      break;
    case 7:  // softmax over 8 head weights
      if (idx == 0) {
        float m = -1e30f;
        for (int i = 0; i < H_; ++i) m = fmaxf(m, hwin[i]);
        float e[H_], s = 0.0f;
        for (int i = 0; i < H_; ++i) { e[i] = __expf(hwin[i] - m); s += e[i]; }
        for (int i = 0; i < H_; ++i) hsoft[i] = e[i] / s;
      }
      break;
  }
}

// ---------------------------------------------------------------------------
// Kernel 2: QKV projection GEMM.  One 16x16 output tile per wave.
// q written pre-scaled by 1/sqrt(HD); v written transposed per head [B,H,32,N].
// ---------------------------------------------------------------------------
__global__ __launch_bounds__(128) void ipa_qkv_gemm(
    const _Float16* __restrict__ x_h, const _Float16* __restrict__ wt,
    const float* __restrict__ bq, const float* __restrict__ bk, const float* __restrict__ bv,
    _Float16* __restrict__ q_h, _Float16* __restrict__ k_h, _Float16* __restrict__ vT) {
  int wave = threadIdx.x >> 5, lane = threadIdx.x & 31;
  long tile = (long)blockIdx.x * 4 + wave;          // 3 * 256 * 16 tiles
  int w  = (int)(tile / (256 * 16));
  int rc = (int)(tile % (256 * 16));
  int row0 = (rc >> 4) * 16, col0 = (rc & 15) * 16;

  const _Float16* A  = x_h + (long)row0 * D_;
  const _Float16* Bm = wt + (long)w * D_ * D_ + (long)col0 * D_;
  v8f acc = {};
#pragma unroll
  for (int kk = 0; kk < D_; kk += 32) {
    __builtin_prefetch(A + kk + 32, 0, 3);
    v16h a = load_frag(A + kk, D_);
    v16h b = load_frag(Bm + kk, D_);
    acc = wmma_f16(a, b, acc);
  }

  int col = col0 + (lane & 15);
  int hi  = lane >> 4;
  const float* bias = (w == 0) ? bq : (w == 1) ? bk : bv;
  float bcol = bias[col];
  const float QSCALE = 0.17677669529663687f;  // 1/sqrt(32)
#pragma unroll
  for (int r = 0; r < 8; ++r) {
    int row = row0 + hi * 8 + r;
    float val = acc[r] + bcol;
    if (w == 0) {
      q_h[(long)row * D_ + col] = (_Float16)(val * QSCALE);
    } else if (w == 1) {
      k_h[(long)row * D_ + col] = (_Float16)val;
    } else {
      int b_ = row >> 11, i = row & (N_ - 1);
      int h = col >> 5, d = col & 31;
      vT[(((long)b_ * H_ + h) * HD_ + d) * N_ + i] = (_Float16)val;
    }
  }
}

// ---------------------------------------------------------------------------
// Kernel 3: fused attention.  One wave = 16 query rows of one (b,h).
// Flash-style streaming over 32-key blocks with online softmax.
// ---------------------------------------------------------------------------
__global__ __launch_bounds__(128) void ipa_attention(
    const _Float16* __restrict__ q_h, const _Float16* __restrict__ k_h,
    const _Float16* __restrict__ vT, const _Float16* __restrict__ tT,
    const float* __restrict__ trans, const unsigned char* __restrict__ mask,
    const float* __restrict__ hsoft, _Float16* __restrict__ feat) {
  __shared__ _Float16 lds_p[4][16 * 32];
  int wave = threadIdx.x >> 5, lane = threadIdx.x & 31;
  int hi = lane >> 4, col = lane & 15;
  int b = blockIdx.z, h = blockIdx.y;
  int row0 = (blockIdx.x * 4 + wave) * 16;
  const float hw = hsoft[h];

  // Q fragment (16 rows x full head dim 32), already scaled by 1/sqrt(hd)
  v16h qf = load_frag(q_h + ((long)b * N_ + row0) * D_ + h * HD_, D_);

  float ti[8][3];
#pragma unroll
  for (int r = 0; r < 8; ++r) {
    const float* tp = trans + ((long)b * N_ + row0 + hi * 8 + r) * 3;
    ti[r][0] = tp[0]; ti[r][1] = tp[1]; ti[r][2] = tp[2];
  }

  float m[8], l[8], dsum[8];
#pragma unroll
  for (int r = 0; r < 8; ++r) { m[r] = -1e30f; l[r] = 0.0f; dsum[r] = 0.0f; }
  v8f acc0 = {}, acc1 = {}, tacc = {};
  _Float16* pslab = &lds_p[wave][0];

  for (int jb = 0; jb < N_; jb += 32) {
    v16h kb0 = load_frag(k_h + ((long)b * N_ + jb) * D_ + h * HD_, D_);
    v16h kb1 = load_frag(k_h + ((long)b * N_ + jb + 16) * D_ + h * HD_, D_);
    v8f z = {};
    v8f s0 = wmma_f16(qf, kb0, z);   // columns j = jb + 0..15
    v8f s1 = wmma_f16(qf, kb1, z);   // columns j = jb + 16..31

    int j0 = jb + col, j1 = j0 + 16;
    const float* tj0 = trans + ((long)b * N_ + j0) * 3;
    const float* tj1 = trans + ((long)b * N_ + j1) * 3;
    float a0 = tj0[0], a1 = tj0[1], a2 = tj0[2];
    float c0 = tj1[0], c1 = tj1[1], c2 = tj1[2];
    bool mk0 = mask[(long)b * N_ + j0] != 0;
    bool mk1 = mask[(long)b * N_ + j1] != 0;

    float p0[8], p1[8];
#pragma unroll
    for (int r = 0; r < 8; ++r) {
      float dx = ti[r][0] - a0, dy = ti[r][1] - a1, dz = ti[r][2] - a2;
      float d0 = dx * dx + dy * dy + dz * dz;
      dx = ti[r][0] - c0; dy = ti[r][1] - c1; dz = ti[r][2] - c2;
      float d1 = dx * dx + dy * dy + dz * dz;
      dsum[r] += d0 + d1;
      float v0 = s0[r] - 0.5f * d0 * hw;
      float v1 = s1[r] - 0.5f * d1 * hw;
      v0 = mk0 ? v0 : -1e9f;
      v1 = mk1 ? v1 : -1e9f;
      // row max across the 16 lanes of this half (rows split at lane 16)
      float mx = fmaxf(v0, v1);
      mx = fmaxf(mx, __shfl_xor(mx, 1, 32));
      mx = fmaxf(mx, __shfl_xor(mx, 2, 32));
      mx = fmaxf(mx, __shfl_xor(mx, 4, 32));
      mx = fmaxf(mx, __shfl_xor(mx, 8, 32));
      float mn = fmaxf(m[r], mx);
      float alpha = __expf(m[r] - mn);
      m[r] = mn;
      float e0 = __expf(v0 - mn), e1 = __expf(v1 - mn);
      float rs = e0 + e1;
      rs += __shfl_xor(rs, 1, 32);
      rs += __shfl_xor(rs, 2, 32);
      rs += __shfl_xor(rs, 4, 32);
      rs += __shfl_xor(rs, 8, 32);
      l[r] = l[r] * alpha + rs;
      acc0[r] *= alpha; acc1[r] *= alpha; tacc[r] *= alpha;
      p0[r] = e0; p1[r] = e1;
    }

    // Stage P (16x32 f16) through LDS to re-shape C-layout -> A-fragment
#pragma unroll
    for (int r = 0; r < 8; ++r) {
      int prow = hi * 8 + r;
      pslab[prow * 32 + col]      = (_Float16)p0[r];
      pslab[prow * 32 + 16 + col] = (_Float16)p1[r];
    }
    asm volatile("s_wait_dscnt 0" ::: "memory");
    v16h pa  = load_frag(pslab, 32);
    v16h vb0 = load_frag(vT + (((long)b * H_ + h) * HD_ + 0)  * N_ + jb, N_);
    v16h vb1 = load_frag(vT + (((long)b * H_ + h) * HD_ + 16) * N_ + jb, N_);
    v16h tb  = load_frag_pred(tT + (long)b * 3 * N_ + jb, N_, col < 3);
    acc0 = wmma_f16(pa, vb0, acc0);   // out dims d = 0..15
    acc1 = wmma_f16(pa, vb1, acc1);   // out dims d = 16..31
    tacc = wmma_f16(pa, tb, tacc);    // weighted translations (cols 0..2)
  }

  // Epilogue: normalize, subtract own translation, mean distance; write feat
#pragma unroll
  for (int r = 0; r < 8; ++r) {
    int i = row0 + hi * 8 + r;
    float inv = 1.0f / l[r];
    long fr = ((long)b * N_ + i) * KPAD_;
    feat[fr + h * HD_ + col]      = (_Float16)(acc0[r] * inv);
    feat[fr + h * HD_ + 16 + col] = (_Float16)(acc1[r] * inv);
    if (col < 3)
      feat[fr + D_ + h * 3 + col] = (_Float16)(tacc[r] * inv - ti[r][col]);
    float ds = dsum[r];
    ds += __shfl_xor(ds, 1, 32);
    ds += __shfl_xor(ds, 2, 32);
    ds += __shfl_xor(ds, 4, 32);
    ds += __shfl_xor(ds, 8, 32);
    if (col == 0)
      feat[fr + D_ + 192 + h] = (_Float16)sqrtf(ds * (1.0f / (float)N_) + 1e-8f);
  }
}

// ---------------------------------------------------------------------------
// Kernel 4: output projection GEMM  out = feat[4096,480] @ Wo_t^T + bo (f32)
// ---------------------------------------------------------------------------
__global__ __launch_bounds__(128) void ipa_out_gemm(
    const _Float16* __restrict__ feat, const _Float16* __restrict__ wo_t,
    const float* __restrict__ bo, float* __restrict__ out) {
  int wave = threadIdx.x >> 5, lane = threadIdx.x & 31;
  long tile = (long)blockIdx.x * 4 + wave;          // 256 row tiles x 16 col tiles
  int row0 = (int)(tile >> 4) * 16, col0 = (int)(tile & 15) * 16;

  const _Float16* A  = feat + (long)row0 * KPAD_;
  const _Float16* Bm = wo_t + (long)col0 * KPAD_;
  v8f acc = {};
#pragma unroll
  for (int kk = 0; kk < KPAD_; kk += 32) {
    __builtin_prefetch(A + kk + 32, 0, 3);
    v16h a = load_frag(A + kk, KPAD_);
    v16h b = load_frag(Bm + kk, KPAD_);
    acc = wmma_f16(a, b, acc);
  }
  int col = col0 + (lane & 15);
  int hi = lane >> 4;
  float bc = bo[col];
#pragma unroll
  for (int r = 0; r < 8; ++r) {
    int row = row0 + hi * 8 + r;
    out[(long)row * D_ + col] = acc[r] + bc;
  }
}

// ---------------------------------------------------------------------------
extern "C" void kernel_launch(void* const* d_in, const int* in_sizes, int n_in,
                              void* d_out, int out_size, void* d_ws, size_t ws_size,
                              hipStream_t stream) {
  const float* x     = (const float*)d_in[0];
  // d_in[1] = rotations: dead code in reference, unused
  const float* trans = (const float*)d_in[2];
  const unsigned char* mask = (const unsigned char*)d_in[3];
  const float* Wq = (const float*)d_in[4];
  const float* bq = (const float*)d_in[5];
  const float* Wk = (const float*)d_in[6];
  const float* bk = (const float*)d_in[7];
  const float* Wv = (const float*)d_in[8];
  const float* bv = (const float*)d_in[9];
  const float* Wo = (const float*)d_in[10];
  const float* bo = (const float*)d_in[11];
  const float* hwin = (const float*)d_in[12];
  float* out = (float*)d_out;

  char* ws = (char*)d_ws;
  size_t off = 0;
  auto alloc = [&](size_t bytes) -> char* {
    char* p = ws + off;
    off += (bytes + 255) & ~(size_t)255;
    return p;
  };
  _Float16* x_h  = (_Float16*)alloc((size_t)BN_ * D_ * 2);        // 2 MB
  _Float16* wt   = (_Float16*)alloc((size_t)3 * D_ * D_ * 2);     // 384 KB
  _Float16* wo_t = (_Float16*)alloc((size_t)D_ * KPAD_ * 2);      // 240 KB
  _Float16* tT   = (_Float16*)alloc((size_t)B_ * 3 * N_ * 2);     // 24 KB
  _Float16* q_h  = (_Float16*)alloc((size_t)BN_ * D_ * 2);        // 2 MB
  _Float16* k_h  = (_Float16*)alloc((size_t)BN_ * D_ * 2);        // 2 MB
  _Float16* vT   = (_Float16*)alloc((size_t)B_ * H_ * HD_ * N_ * 2); // 2 MB
  _Float16* feat = (_Float16*)alloc((size_t)BN_ * KPAD_ * 2);     // 3.75 MB
  float*    hs   = (float*)alloc(H_ * sizeof(float));

  // 1) prep: conversions / transposes / zero-fill / head softmax
  {
    dim3 grid(((long)BN_ * KPAD_ + 255) / 256, 8);
    ipa_prep<<<grid, 256, 0, stream>>>(x, trans, Wq, Wk, Wv, Wo, hwin,
                                       x_h, wt, wo_t, tT, feat, hs);
  }
  // 2) QKV projections (3 * 256 * 16 tiles, 4 waves/block)
  ipa_qkv_gemm<<<3 * 256 * 16 / 4, 128, 0, stream>>>(x_h, wt, bq, bk, bv,
                                                     q_h, k_h, vT);
  // 3) fused attention: grid (N/16/4, H, B)
  {
    dim3 grid(N_ / 16 / 4, H_, B_);
    ipa_attention<<<grid, 128, 0, stream>>>(q_h, k_h, vT, tT, trans, mask, hs, feat);
  }
  // 4) output projection (256 * 16 tiles, 4 waves/block)
  ipa_out_gemm<<<256 * 16 / 4, 128, 0, stream>>>(feat, wo_t, bo, out);
}
